// GCNEncoder_22728966930847
// MI455X (gfx1250) — matrix operand order repaired and checked by reading, not compiled, not run on hardware
//
#include <hip/hip_runtime.h>
#include <hip/hip_bf16.h>

// ---------------------------------------------------------------------------
// GCN 2-layer forward for MI455X (gfx1250).
//  - GEMMs via v_wmma_f32_16x16x32_bf16 with 3-term bf16 split (fp32-quality)
//  - weights pre-split (hi/lo) + pre-swizzled into WMMA B-fragment layout once
//  - scatter-add aggregation via native fp32 global atomics (L2-resident)
//  - BN train stats via two-stage reduction
// ---------------------------------------------------------------------------

typedef __attribute__((ext_vector_type(16))) __bf16 v16bf;
typedef __attribute__((ext_vector_type(8)))  float  v8f;

#define IN_CH  128
#define HID_CH 256
#define OUT_CH 128
#define BN_EPS 1e-5f

__device__ __forceinline__ void atomAddF(float* p, float v) {
  // lowers to global_atomic_add_f32 (no CAS loop) on gfx1250
  unsafeAtomicAdd(p, v);
}

// ---- degree: deg[dst[e]] += 1 ---------------------------------------------
__global__ void k_degree(const int* __restrict__ dst, unsigned* __restrict__ deg, int E) {
  int e = blockIdx.x * blockDim.x + threadIdx.x;
  if (e < E) atomicAdd(&deg[dst[e]], 1u);
}

// ---- dis[i] = rsqrt(deg[i] + 1) -------------------------------------------
__global__ void k_dis(const unsigned* __restrict__ deg, float* __restrict__ dis, int N) {
  int i = blockIdx.x * blockDim.x + threadIdx.x;
  if (i < N) dis[i] = rsqrtf((float)deg[i] + 1.0f);
}

// ---- pre-split + pre-swizzle W[K,N] into WMMA B-fragment layout ------------
// fragment f = ks*(N/16)+ct holds the 32x16 bf16 tile for K-step ks, col-tile ct.
// lane = hsel*16 + (n&15) with hsel = (k>>4)&1; element e = k&15.
// stored contiguously: off = (f*32 + lane)*16 + e   (bf16 units)
__global__ void k_prepW(const float* __restrict__ W, __bf16* __restrict__ hi,
                        __bf16* __restrict__ lo, int K, int N)
{
  int t = blockIdx.x * blockDim.x + threadIdx.x;
  if (t >= K * N) return;
  int k = t / N, n = t - (t / N) * N;
  float w = W[t];
  __bf16 h = (__bf16)w;
  __bf16 l = (__bf16)(w - (float)h);
  int ks   = k >> 5;
  int hsel = (k >> 4) & 1;
  int ct   = n >> 4;
  int tiles_n = N >> 4;
  size_t off = ((size_t)(ks * tiles_n + ct) * 32 + hsel * 16 + (n & 15)) * 16 + (k & 15);
  hi[off] = h;
  lo[off] = l;
}

// ---- C[M,N] = A[M,K] @ B[K,N], fp32 A/C, pre-swizzled bf16 B --------------
// one 16x16 output tile per wave, 8 waves per block. M%16==0, N%16==0, K%32==0
__global__ void __launch_bounds__(256)
k_gemm_bf16x3(const float* __restrict__ A, const v16bf* __restrict__ Bhi,
              const v16bf* __restrict__ Blo, float* __restrict__ C,
              int M, int K, int N)
{
  const int lane = threadIdx.x & 31;
  const int wave = threadIdx.x >> 5;
  const int tiles_n = N >> 4;
  const int total_tiles = (M >> 4) * tiles_n;
  const int tile = blockIdx.x * 8 + wave;
  if (tile >= total_tiles) return;             // wave-uniform: EXEC stays all-1s

  const int tileN   = tile % tiles_n;
  const int rowBase = (tile / tiles_n) << 4;
  const int colBase = tileN << 4;
  const int hs = lane >> 4;                    // lane half-select
  const int ml = lane & 15;
  const int rowA = rowBase + ml;               // A: lane holds row rowA

  v8f acc = {};
  int ks = 0;
  for (int k0 = 0; k0 < K; k0 += 32, ++ks) {
    // A fragment: one 16-wide contiguous K run matching the B pre-swizzle
    //   elements 0..15: K = k0 + hs*16 + (0..15)
    const float4* pa = reinterpret_cast<const float4*>(A + (size_t)rowA * K + k0 + hs * 16);
    float4 a0 = pa[0], a1 = pa[1], a2 = pa[2], a3 = pa[3];
    float av[16] = {a0.x,a0.y,a0.z,a0.w, a1.x,a1.y,a1.z,a1.w,
                    a2.x,a2.y,a2.z,a2.w, a3.x,a3.y,a3.z,a3.w};

    // B fragments: single contiguous 32B load per lane (pre-swizzled)
    size_t bidx = (size_t)(ks * tiles_n + tileN) * 32 + lane;
    v16bf bh = Bhi[bidx];
    v16bf bl = Blo[bidx];

    // split fp32 A -> bf16 hi + lo
    v16bf ah, al;
    #pragma unroll
    for (int e = 0; e < 16; ++e) {
      __bf16 h = (__bf16)av[e];
      ah[e] = h;
      al[e] = (__bf16)(av[e] - (float)h);
    }
    acc = __builtin_amdgcn_wmma_f32_16x16x32_bf16(false, ah, false, bh, (short)0, acc, false, false);
    acc = __builtin_amdgcn_wmma_f32_16x16x32_bf16(false, al, false, bh, (short)0, acc, false, false);
    acc = __builtin_amdgcn_wmma_f32_16x16x32_bf16(false, ah, false, bl, (short)0, acc, false, false);
  }

  // C/D layout: VGPR r -> M = r + hs*8, N = colBase + ml (lane-coalesced)
  #pragma unroll
  for (int e = 0; e < 8; ++e) {
    int m = e + hs * 8;
    C[(size_t)(rowBase + m) * N + colBase + ml] = acc[e];
  }
}

// ---- edge scatter: agg[dst] += h[src] * dis[src]*dis[dst] ------------------
// one thread per (edge, 4-channel chunk); float4 gather, 4 fp32 atomics
__global__ void k_scatter(const float* __restrict__ h, const int* __restrict__ src,
                          const int* __restrict__ dst, const float* __restrict__ dis,
                          float* __restrict__ agg, int E, int C4, int C)
{
  int t = blockIdx.x * blockDim.x + threadIdx.x;
  int e = t / C4;
  if (e >= E) return;
  int ch4 = t - e * C4;
  int s = src[e], d = dst[e];
  float nrm = dis[s] * dis[d];
  const float4* h4 = reinterpret_cast<const float4*>(h);
  float4 v = h4[(size_t)s * C4 + ch4];
  float* ag = agg + (size_t)d * C + ch4 * 4;
  atomAddF(ag + 0, v.x * nrm);
  atomAddF(ag + 1, v.y * nrm);
  atomAddF(ag + 2, v.z * nrm);
  atomAddF(ag + 3, v.w * nrm);
}

// ---- z = agg + h*dis^2 + b (in place over agg) + per-channel BN partials ---
// blockDim == C (one thread per channel), each block reduces `rpb` rows
__global__ void k_finish_stats(const float* __restrict__ agg, const float* __restrict__ h,
                               const float* __restrict__ dis, const float* __restrict__ b,
                               float* __restrict__ z, float* __restrict__ bnsum,
                               float* __restrict__ bnsq, int N, int C, int rpb)
{
  int c  = threadIdx.x;
  int r0 = blockIdx.x * rpb;
  int r1 = min(r0 + rpb, N);
  float bc = b[c];
  float s = 0.f, q = 0.f;
  for (int r = r0; r < r1; ++r) {
    float d = dis[r];
    size_t idx = (size_t)r * C + c;
    float v = agg[idx] + h[idx] * (d * d) + bc;
    z[idx] = v;
    s += v;
    q += v * v;
  }
  atomAddF(&bnsum[c], s);
  atomAddF(&bnsq[c], q);
}

// ---- BN(train) + ReLU ------------------------------------------------------
__global__ void k_bnapply(const float* __restrict__ z, const float* __restrict__ gamma,
                          const float* __restrict__ beta, const float* __restrict__ bnsum,
                          const float* __restrict__ bnsq, float* __restrict__ out,
                          int total, int C, float invN)
{
  int idx = blockIdx.x * blockDim.x + threadIdx.x;
  if (idx >= total) return;
  int c = idx % C;
  float mu  = bnsum[c] * invN;
  float var = bnsq[c] * invN - mu * mu;
  float v = (z[idx] - mu) * rsqrtf(var + BN_EPS) * gamma[c] + beta[c];
  out[idx] = v > 0.f ? v : 0.f;
}

// ---- out = agg + out*dis^2 + b (in place over out) -------------------------
__global__ void k_finish2(const float* __restrict__ agg, const float* __restrict__ dis,
                          const float* __restrict__ b, float* __restrict__ out,
                          int total, int C)
{
  int idx = blockIdx.x * blockDim.x + threadIdx.x;
  if (idx >= total) return;
  int r = idx / C;
  int c = idx - r * C;
  float d = dis[r];
  out[idx] = agg[idx] + out[idx] * (d * d) + b[c];
}

static inline unsigned cdiv(long long a, long long b) { return (unsigned)((a + b - 1) / b); }

extern "C" void kernel_launch(void* const* d_in, const int* in_sizes, int n_in,
                              void* d_out, int out_size, void* d_ws, size_t ws_size,
                              hipStream_t stream)
{
  const float* x   = (const float*)d_in[0];
  const int*   ei  = (const int*)  d_in[1];
  const float* W1  = (const float*)d_in[2];
  const float* b1  = (const float*)d_in[3];
  const float* g1  = (const float*)d_in[4];
  const float* be1 = (const float*)d_in[5];
  const float* W2  = (const float*)d_in[6];
  const float* b2  = (const float*)d_in[7];
  float* out = (float*)d_out;

  const int N = in_sizes[0] / IN_CH;   // 50000
  const int E = in_sizes[1] / 2;       // 800000
  const int* src = ei;
  const int* dst = ei + E;

  // ---- workspace layout (64B-aligned chunks) ----
  char* p = (char*)d_ws;
  auto take = [&](size_t bytes) {
    char* r = p;
    p += (bytes + 63) & ~(size_t)63;
    return r;
  };
  unsigned* deg   = (unsigned*)take(sizeof(unsigned) * (size_t)N);
  float*    dis   = (float*)   take(sizeof(float) * (size_t)N);
  float*    bnsum = (float*)   take(sizeof(float) * HID_CH);
  float*    bnsq  = (float*)   take(sizeof(float) * HID_CH);
  __bf16*   w1hi  = (__bf16*)  take(sizeof(__bf16) * IN_CH  * HID_CH);
  __bf16*   w1lo  = (__bf16*)  take(sizeof(__bf16) * IN_CH  * HID_CH);
  __bf16*   w2hi  = (__bf16*)  take(sizeof(__bf16) * HID_CH * OUT_CH);
  __bf16*   w2lo  = (__bf16*)  take(sizeof(__bf16) * HID_CH * OUT_CH);
  float*    h1    = (float*)   take(sizeof(float) * (size_t)N * HID_CH);
  float*    agg   = (float*)   take(sizeof(float) * (size_t)N * HID_CH);

  hipMemsetAsync(deg,   0, sizeof(unsigned) * (size_t)N, stream);
  hipMemsetAsync(bnsum, 0, sizeof(float) * HID_CH, stream);
  hipMemsetAsync(bnsq,  0, sizeof(float) * HID_CH, stream);
  hipMemsetAsync(agg,   0, sizeof(float) * (size_t)N * HID_CH, stream);

  // degrees + normalization factors
  k_degree<<<cdiv(E, 256), 256, 0, stream>>>(dst, deg, E);
  k_dis<<<cdiv(N, 256), 256, 0, stream>>>(deg, dis, N);

  // pre-split/swizzle weights into WMMA B-fragment layout (tiny, one-shot)
  k_prepW<<<cdiv(IN_CH  * HID_CH, 256), 256, 0, stream>>>(W1, w1hi, w1lo, IN_CH,  HID_CH);
  k_prepW<<<cdiv(HID_CH * OUT_CH, 256), 256, 0, stream>>>(W2, w2hi, w2lo, HID_CH, OUT_CH);

  // h1 = x @ W1   [N,128]@[128,256]
  {
    int tiles = (N / 16) * (HID_CH / 16);
    k_gemm_bf16x3<<<cdiv(tiles, 8), 256, 0, stream>>>(
        x, (const v16bf*)w1hi, (const v16bf*)w1lo, h1, N, IN_CH, HID_CH);
  }
  // agg1 += h1[src]*norm  (scatter to dst)
  k_scatter<<<cdiv((long long)E * (HID_CH / 4), 256), 256, 0, stream>>>(
      h1, src, dst, dis, agg, E, HID_CH / 4, HID_CH);
  // z = agg1 + h1*dis^2 + b1 (in place) + BN partial sums
  k_finish_stats<<<cdiv(N, 128), HID_CH, 0, stream>>>(
      agg, h1, dis, b1, agg, bnsum, bnsq, N, HID_CH, 128);
  // hbn = relu(BN(z)) -> reuse h1
  k_bnapply<<<cdiv((long long)N * HID_CH, 256), 256, 0, stream>>>(
      agg, g1, be1, bnsum, bnsq, h1, N * HID_CH, HID_CH, 1.0f / (float)N);

  // h2 = hbn @ W2 -> d_out   [N,256]@[256,128]
  {
    int tiles = (N / 16) * (OUT_CH / 16);
    k_gemm_bf16x3<<<cdiv(tiles, 8), 256, 0, stream>>>(
        h1, (const v16bf*)w2hi, (const v16bf*)w2lo, out, N, HID_CH, OUT_CH);
  }
  // layer-2 aggregation (reuse agg buffer)
  hipMemsetAsync(agg, 0, sizeof(float) * (size_t)N * OUT_CH, stream);
  k_scatter<<<cdiv((long long)E * (OUT_CH / 4), 256), 256, 0, stream>>>(
      out, src, dst, dis, agg, E, OUT_CH / 4, OUT_CH);
  // out = agg2 + h2*dis^2 + b2 (in place)
  k_finish2<<<cdiv((long long)N * OUT_CH, 256), 256, 0, stream>>>(
      agg, dis, b2, out, N * OUT_CH, OUT_CH);
}
